// _MultiHeadAttention_65687229825531
// MI455X (gfx1250) — matrix operand order repaired and checked by reading, not compiled, not run on hardware
//
#include <hip/hip_runtime.h>

typedef __attribute__((ext_vector_type(16))) __bf16 bf16x16;
typedef __attribute__((ext_vector_type(8)))  __bf16 bf16x8;
typedef __attribute__((ext_vector_type(8)))  float  f32x8;

// ---------- helpers ----------

__device__ __forceinline__ __bf16 f2bf(float f) {
  unsigned u = __builtin_bit_cast(unsigned, f);
  u += 0x7FFFu + ((u >> 16) & 1u);           // round-to-nearest-even
  unsigned short h = (unsigned short)(u >> 16);
  return __builtin_bit_cast(__bf16, h);
}

__device__ __forceinline__ f32x8 wmma_bf16(bf16x16 a, bf16x16 b, f32x8 c) {
  return __builtin_amdgcn_wmma_f32_16x16x32_bf16(
      /*neg_a=*/false, a, /*neg_b=*/false, b,
      /*c_mod=*/(short)0, c, /*reuse_a=*/false, /*reuse_b=*/false);
}

// A fragment (16x32, MxK), src row-major with leading dim ld.
// lane L: M=L%16, half=L/16; elems j -> K=half*16+j
__device__ __forceinline__ bf16x16 load_a_frag(const __bf16* tile, int ld) {
  const int lane = threadIdx.x & 31;
  const int m = lane & 15, h = lane >> 4;
  const __bf16* p = tile + (size_t)m * ld + h * 16;
  bf16x8 lo = *(const bf16x8*)(p);
  bf16x8 hi = *(const bf16x8*)(p + 8);
  return __builtin_shufflevector(lo, hi, 0,1,2,3,4,5,6,7,8,9,10,11,12,13,14,15);
}

// B fragment (32x16, KxN), src stored transposed: row n = logical column N,
// row-major over K with leading dim ld. lane L: N=L%16, elems j -> K=(L/16)*16+j
__device__ __forceinline__ bf16x16 load_b_frag(const __bf16* tileT, int ld) {
  const int lane = threadIdx.x & 31;
  const int n = lane & 15, h = lane >> 4;
  return *(const bf16x16*)(tileT + (size_t)n * ld + h * 16);
}

// ---------- conversion / layout kernels ----------

__global__ void __launch_bounds__(256) k_cvt_bf16(const float* __restrict__ src,
                                                  __bf16* __restrict__ dst, int n) {
  int i = blockIdx.x * 256 + threadIdx.x;
  if (i < n) dst[i] = f2bf(src[i]);
}

// Wqkv^T : [3072][2048] bf16, row n = column n of [Wq|Wk|Wv]
__global__ void __launch_bounds__(256) k_build_wqkvT(const float* __restrict__ Wq,
                                                     const float* __restrict__ Wk,
                                                     const float* __restrict__ Wv,
                                                     __bf16* __restrict__ out) {
  int i = blockIdx.x * 256 + threadIdx.x;     // over 3072*2048
  int k = i & 2047, n = i >> 11;
  float v;
  if (n < 2048)      v = Wq[(size_t)k * 2048 + n];
  else if (n < 2560) v = Wk[(size_t)k * 512 + (n - 2048)];
  else               v = Wv[(size_t)k * 512 + (n - 2560)];
  out[i] = f2bf(v);
}

__global__ void __launch_bounds__(256) k_build_woT(const float* __restrict__ Wo,
                                                   __bf16* __restrict__ out) {
  int i = blockIdx.x * 256 + threadIdx.x;     // over 2048*2048
  int k = i & 2047, n = i >> 11;
  out[i] = f2bf(Wo[(size_t)k * 2048 + n]);
}

// Vt[b][g][d(128)][t(2048)] = qkv[b*T+t][2560 + g*128 + d]
__global__ void __launch_bounds__(256) k_transpose_v(const __bf16* __restrict__ qkv,
                                                     __bf16* __restrict__ vt) {
  int i = blockIdx.x * 256 + threadIdx.x;     // over 2*4*128*2048
  int t = i & 2047; int r = i >> 11;
  int d = r & 127; r >>= 7;
  int g = r & 3;   int b = r >> 2;
  vt[i] = qkv[((size_t)(b * 2048) + t) * 3072 + 2560 + g * 128 + d];
}

// ---------- bf16 WMMA GEMM: C[M,N] = A[M,K] @ Bt[N,K]^T ----------
// block = 256 thr = 8 waves in a 4(M) x 2(N) grid; wave tile = 32x64
// (2 A frags x 4 B frags -> 8 WMMAs per K32 step, B reused across M)

template <bool OUT_F32>
__global__ void __launch_bounds__(256) k_gemm_bf16(const __bf16* __restrict__ A,
                                                   const __bf16* __restrict__ Bt,
                                                   void* __restrict__ Cout,
                                                   int M, int N, int K) {
  const int w = threadIdx.x >> 5;
  const int lane = threadIdx.x & 31;
  const int wm = w >> 1, wn = w & 1;
  const int bm = blockIdx.x * 128 + wm * 32;
  const int bn = blockIdx.y * 128 + wn * 64;

  f32x8 c[2][4];
  #pragma unroll
  for (int r = 0; r < 2; ++r)
    #pragma unroll
    for (int t = 0; t < 4; ++t) c[r][t] = (f32x8){0,0,0,0,0,0,0,0};

  const __bf16* arow0 = A + (size_t)bm * K;
  const __bf16* arow1 = arow0 + (size_t)16 * K;

  for (int k0 = 0; k0 < K; k0 += 32) {
    // prefetch 4 K-steps ahead (global_prefetch_b8): A rows + all 64 B rows
    if (k0 + 128 < K) {
      __builtin_prefetch(arow0 + (size_t)(lane & 15) * K + k0 + 128, 0, 3);
      __builtin_prefetch(arow1 + (size_t)(lane & 15) * K + k0 + 128, 0, 3);
      __builtin_prefetch(Bt + (size_t)(bn + lane) * K + k0 + 128, 0, 3);
      __builtin_prefetch(Bt + (size_t)(bn + 32 + lane) * K + k0 + 128, 0, 3);
    }
    bf16x16 a0 = load_a_frag(arow0 + k0, K);
    bf16x16 a1 = load_a_frag(arow1 + k0, K);
    #pragma unroll
    for (int t = 0; t < 4; ++t) {
      bf16x16 b = load_b_frag(Bt + (size_t)(bn + t * 16) * K + k0, K);
      c[0][t] = wmma_bf16(a0, b, c[0][t]);
      c[1][t] = wmma_bf16(a1, b, c[1][t]);
    }
  }

  const int n0 = lane & 15, h = lane >> 4;
  #pragma unroll
  for (int r = 0; r < 2; ++r) {
    #pragma unroll
    for (int t = 0; t < 4; ++t) {
      #pragma unroll
      for (int i = 0; i < 8; ++i) {
        size_t row = (size_t)(bm + r * 16 + i + 8 * h);
        size_t col = (size_t)(bn + t * 16 + n0);
        if (OUT_F32) ((float*)Cout)[row * N + col] = c[r][t][i];
        else         ((__bf16*)Cout)[row * N + col] = f2bf(c[r][t][i]);
      }
    }
  }
}

// ---------- flash attention (causal, GQA) ----------
// one wave per (b, head, 16-row q tile); T=2048, hd=128, 3072 = qkv row stride

__global__ void __launch_bounds__(256) k_flash_attn(const __bf16* __restrict__ qkv,
                                                    const __bf16* __restrict__ vt,
                                                    __bf16* __restrict__ attn) {
  __shared__ __bf16 ldsP[8][16][32];          // per-wave 16x32 P tile (bf16)
  const int w = threadIdx.x >> 5;
  const int lane = threadIdx.x & 31;
  const int n0 = lane & 15, h = lane >> 4;

  int idx = blockIdx.x * 8 + w;
  const int qt = idx & 127; idx >>= 7;        // 128 q tiles
  const int hd = idx & 15;  const int b = idx >> 4;
  const int g = hd >> 2;                      // R = 4
  const float scale = 0.08838834764831845f;   // 1/sqrt(128)
  const float NEG_INF = -__builtin_inff();

  // resident Q fragments: 16 rows x 128 (4 x K32 chunks)
  const __bf16* qbase = qkv + ((size_t)(b * 2048) + qt * 16) * 3072 + hd * 128;
  bf16x16 aq[4];
  #pragma unroll
  for (int cc = 0; cc < 4; ++cc) aq[cc] = load_a_frag(qbase + cc * 32, 3072);

  const __bf16* kbase = qkv + (size_t)(b * 2048) * 3072 + 2048 + g * 128;
  const __bf16* vtb   = vt + (size_t)(b * 4 + g) * 128 * 2048;

  f32x8 o[8];
  #pragma unroll
  for (int d = 0; d < 8; ++d) o[d] = (f32x8){0,0,0,0,0,0,0,0};
  float m_i[8], l_i[8];
  #pragma unroll
  for (int i = 0; i < 8; ++i) { m_i[i] = NEG_INF; l_i[i] = 0.f; }

  const int kend = qt * 16 + 16;              // causal bound (exclusive)
  for (int j0 = 0; j0 < kend; j0 += 32) {
    // ---- S = Q K^T for 32 keys (two 16x16 tiles) ----
    f32x8 s0 = (f32x8){0,0,0,0,0,0,0,0};
    f32x8 s1 = (f32x8){0,0,0,0,0,0,0,0};
    #pragma unroll
    for (int cc = 0; cc < 4; ++cc) {
      s0 = wmma_bf16(aq[cc], load_b_frag(kbase + (size_t)(j0) * 3072 + cc * 32, 3072), s0);
      s1 = wmma_bf16(aq[cc], load_b_frag(kbase + (size_t)(j0 + 16) * 3072 + cc * 32, 3072), s1);
    }

    // ---- mask + online softmax (row M = i + 8*h lives on one 16-lane half) ----
    float p0[8], p1[8], alpha[8];
    #pragma unroll
    for (int i = 0; i < 8; ++i) {
      int qrow = qt * 16 + i + 8 * h;
      int k0i = j0 + n0, k1i = j0 + 16 + n0;
      float v0 = (k0i <= qrow) ? s0[i] * scale : NEG_INF;
      float v1 = (k1i <= qrow) ? s1[i] * scale : NEG_INF;
      float rmax = fmaxf(v0, v1);
      rmax = fmaxf(rmax, __shfl_xor(rmax, 1));
      rmax = fmaxf(rmax, __shfl_xor(rmax, 2));
      rmax = fmaxf(rmax, __shfl_xor(rmax, 4));
      rmax = fmaxf(rmax, __shfl_xor(rmax, 8));
      float mnew = fmaxf(m_i[i], rmax);       // always finite (diag present)
      float a = __expf(m_i[i] - mnew);        // 0 on first chunk
      p0[i] = __expf(v0 - mnew);
      p1[i] = __expf(v1 - mnew);
      float rs = p0[i] + p1[i];
      rs += __shfl_xor(rs, 1);
      rs += __shfl_xor(rs, 2);
      rs += __shfl_xor(rs, 4);
      rs += __shfl_xor(rs, 8);
      l_i[i] = l_i[i] * a + rs;
      m_i[i] = mnew;
      alpha[i] = a;
    }

    // ---- C-layout P -> LDS -> A-layout fragment ----
    #pragma unroll
    for (int i = 0; i < 8; ++i) {
      ldsP[w][i + 8 * h][n0]      = f2bf(p0[i]);
      ldsP[w][i + 8 * h][16 + n0] = f2bf(p1[i]);
    }
    asm volatile("s_wait_dscnt 0" ::: "memory");   // DS in-order per wave; fence compiler too
    bf16x16 pa = load_a_frag(&ldsP[w][0][0], 32);
    asm volatile("" ::: "memory");                 // keep next-iter stores after these loads

    // ---- O = diag(alpha) O + P V ----
    #pragma unroll
    for (int d = 0; d < 8; ++d) {
      #pragma unroll
      for (int i = 0; i < 8; ++i) o[d][i] *= alpha[i];
      bf16x16 vb = load_b_frag(vtb + (size_t)(d * 16) * 2048 + j0, 2048);
      o[d] = wmma_bf16(pa, vb, o[d]);
    }
  }

  // ---- normalize + store bf16 attn [B*T][2048] ----
  float inv_l[8];
  #pragma unroll
  for (int i = 0; i < 8; ++i) inv_l[i] = 1.0f / l_i[i];
  __bf16* obase = attn + ((size_t)(b * 2048) + qt * 16) * 2048 + hd * 128;
  #pragma unroll
  for (int d = 0; d < 8; ++d) {
    #pragma unroll
    for (int i = 0; i < 8; ++i)
      obase[(size_t)(i + 8 * h) * 2048 + d * 16 + n0] = f2bf(o[d][i] * inv_l[i]);
  }
}

// ---------- launch ----------

extern "C" void kernel_launch(void* const* d_in, const int* in_sizes, int n_in,
                              void* d_out, int out_size, void* d_ws, size_t ws_size,
                              hipStream_t stream) {
  (void)in_sizes; (void)n_in; (void)out_size; (void)ws_size;
  const float* x  = (const float*)d_in[0];   // [2,2048,2048]
  const float* Wq = (const float*)d_in[1];   // [2048,2048]
  const float* Wk = (const float*)d_in[2];   // [2048,512]
  const float* Wv = (const float*)d_in[3];   // [2048,512]
  const float* Wo = (const float*)d_in[4];   // [2048,2048]
  float* out = (float*)d_out;                // [2,2048,2048] f32

  char* ws = (char*)d_ws;
  __bf16* xb     = (__bf16*)(ws);                         // 4096*2048      = 16.8 MB
  __bf16* wqkvT  = (__bf16*)(ws + 16777216);              // 3072*2048      = 12.6 MB
  __bf16* woT    = (__bf16*)(ws + 29360128);              // 2048*2048      =  8.4 MB
  __bf16* qkv    = (__bf16*)(ws + 37748736);              // 4096*3072      = 25.2 MB
  __bf16* vtt    = (__bf16*)(ws + 62914560);              // 2*4*128*2048   =  4.2 MB
  __bf16* attn   = (__bf16*)(ws + 67108864);              // 4096*2048      = 16.8 MB
                                                          // total 83.9 MB

  // 1) precision conversion + weight transposes
  k_cvt_bf16<<<(4096 * 2048) / 256, 256, 0, stream>>>(x, xb, 4096 * 2048);
  k_build_wqkvT<<<(3072 * 2048) / 256, 256, 0, stream>>>(Wq, Wk, Wv, wqkvT);
  k_build_woT<<<(2048 * 2048) / 256, 256, 0, stream>>>(Wo, woT);

  // 2) fused QKV projection: [4096,2048] @ [2048,3072] -> bf16 [4096,3072]
  k_gemm_bf16<false><<<dim3(4096 / 128, 3072 / 128), 256, 0, stream>>>(
      xb, wqkvT, qkv, 4096, 3072, 2048);

  // 3) V transpose for contiguous PV B-fragments
  k_transpose_v<<<(2 * 4 * 128 * 2048) / 256, 256, 0, stream>>>(qkv, vtt);

  // 4) causal GQA flash attention -> bf16 [4096,2048]
  k_flash_attn<<<(2 * 16 * 128) / 8, 256, 0, stream>>>(qkv, vtt, attn);

  // 5) output projection: [4096,2048] @ [2048,2048] -> f32 d_out
  k_gemm_bf16<true><<<dim3(4096 / 128, 2048 / 128), 256, 0, stream>>>(
      attn, woT, out, 4096, 2048, 2048);
}
